// BatchMultiHeadGraphAttention_5171140625169
// MI455X (gfx1250) — matrix operand order repaired and checked by reading, not compiled, not run on hardware
//
#include <hip/hip_runtime.h>
#include <hip/hip_bf16.h>

typedef __attribute__((ext_vector_type(16))) __bf16 v16bf;
typedef __attribute__((ext_vector_type(8)))  float  v8f;

#define BS      8
#define NNODE   1024
#define FIN     256
#define NHEAD   8
#define FOUT    64
#define NEG_SLOPE 0.2f

// ---------------- workspace layout (bytes) ----------------
// hA  : packed bf16 A-operand of h     : BS*(N/16)*8kt*512       = 4 MB
// wB  : packed bf16 B-operand of w     : H*8kt*4ot*512           = 256 KB
// hpB : packed bf16 B-operand of h'    : BS*H*(N/32)*4ot*512     = 8 MB
// s_src, s_dst : [BS*H*N] f32          : 256 KB each
#define OFF_HA   0u
#define OFF_WB   (4u*1024*1024)
#define OFF_HPB  (OFF_WB + 256u*1024)
#define OFF_SSRC (OFF_HPB + 8u*1024*1024)
#define OFF_SDST (OFF_SSRC + 256u*1024)

// ---------------- prep kernels ----------------
__global__ void zero_f32(float* p, int n) {
    int i = blockIdx.x * 256 + threadIdx.x;
    if (i < n) p[i] = 0.0f;
}

// A-operand layout: tile (b, rt16, kt) -> [lane][i], lane holds row rt16*16 + lane%16,
// k = kt*32 + 16*(i/8) + 8*(lane/16) + (i%8)
__global__ void pack_hA(const float* __restrict__ h, __bf16* __restrict__ hA) {
    int idx  = blockIdx.x * 256 + threadIdx.x;       // 2,097,152 total
    int i    = idx & 15;
    int lane = (idx >> 4) & 31;
    int kt   = (idx >> 9) & 7;
    int rt   = (idx >> 12) & 63;
    int b    = idx >> 18;
    int row  = rt * 16 + (lane & 15);
    int k    = kt * 32 + 16 * (i >> 3) + 8 * (lane >> 4) + (i & 7);
    hA[idx] = (__bf16)h[(b * NNODE + row) * FIN + k];
}

// B-operand layout: tile (head, kt, ot) -> [lane][i], lane holds col ot*16 + lane%16,
// k = kt*32 + 16*(i/8) + 8*(lane/16) + (i%8)
__global__ void pack_wB(const float* __restrict__ w, __bf16* __restrict__ wB) {
    int idx  = blockIdx.x * 256 + threadIdx.x;       // 131,072 total
    int i    = idx & 15;
    int lane = (idx >> 4) & 31;
    int ot   = (idx >> 9) & 3;
    int kt   = (idx >> 11) & 7;
    int head = idx >> 14;
    int col  = ot * 16 + (lane & 15);
    int k    = kt * 32 + 16 * (i >> 3) + 8 * (lane >> 4) + (i & 7);
    wB[idx] = (__bf16)w[(head * FIN + k) * FOUT + col];
}

// ---------------- kernel 1: h' = h@w (WMMA), tanh, attn scalars, repack h' ----
__global__ __launch_bounds__(256)
void gat_project(const __bf16* __restrict__ hA, const __bf16* __restrict__ wB,
                 const float* __restrict__ a_src, const float* __restrict__ a_dst,
                 __bf16* __restrict__ hpB, float* __restrict__ s_src,
                 float* __restrict__ s_dst) {
    int blk  = blockIdx.x;                 // BS*NHEAD*(N/32) = 2048
    int rt32 = blk & 31;
    int head = (blk >> 5) & 7;
    int b    = blk >> 8;
    int bh   = b * NHEAD + head;

    int wave = threadIdx.x >> 5;
    int lane = threadIdx.x & 31;
    int sub  = wave >> 2;                  // which 16-row sub-tile
    int ot   = wave & 3;                   // which 16-col tile
    int half = lane >> 4, mrow = lane & 15;
    int rt16 = rt32 * 2 + sub;

    const v16bf* pA = (const v16bf*)hA + (size_t)((b * 64 + rt16) * 8) * 32;
    const v16bf* pB = (const v16bf*)wB + (size_t)((head * 8) * 4 + ot) * 32;

    v8f acc = {0.f, 0.f, 0.f, 0.f, 0.f, 0.f, 0.f, 0.f};
#pragma unroll
    for (int kt = 0; kt < 8; ++kt) {
        v16bf a  = pA[kt * 32 + lane];
        v16bf bb = pB[kt * 4 * 32 + lane];
        acc = __builtin_amdgcn_wmma_f32_16x16x32_bf16(false, a, false, bb,
                                                      (short)0, acc, false, false);
    }

    __shared__ __bf16 tile[32][64];        // h' staged for B-layout repack
    int rowbase = rt32 * 32 + sub * 16;
    int col     = ot * 16 + mrow;
    float av = a_src[head * FOUT + col];
    float dv = a_dst[head * FOUT + col];
#pragma unroll
    for (int r = 0; r < 8; ++r) {
        float hp = acc[r];
        tile[sub * 16 + r + 8 * half][col] = (__bf16)hp;
        float t  = tanhf(hp);
        float vs = t * av;
        float vd = t * dv;
#pragma unroll
        for (int off = 1; off < 16; off <<= 1) {
            vs += __shfl_xor(vs, off, 16);
            vd += __shfl_xor(vd, off, 16);
        }
        if (mrow == 0) {
            int row = rowbase + r + 8 * half;
            atomicAdd(&s_src[bh * NNODE + row], vs);
            atomicAdd(&s_dst[bh * NNODE + row], vd);
        }
    }
    __syncthreads();

    // repack 32x64 tile into attention B-operand layout
    int tid = threadIdx.x;
    if (tid < 128) {
        int lane2 = tid & 31, ot2 = tid >> 5;
        int c2 = ot2 * 16 + (lane2 & 15);
        v16bf v;
#pragma unroll
        for (int i = 0; i < 16; ++i)
            v[i] = tile[16 * (i >> 3) + 8 * (lane2 >> 4) + (i & 7)][c2];
        *(v16bf*)(hpB + (size_t)(((bh * 32 + rt32) * 4 + ot2) * 512) + lane2 * 16) = v;
    }
}

// ---------------- kernel 2: masked softmax (fixed safe shift) + P@h' (WMMA) --
// Exploits monotonicity of leaky_relu: M_i = lrelu(s_i + max_j d_j) is an upper
// bound for every score in row i, so exp(score - M_i) never overflows and the
// accumulators never need rescaling -> pure WMMA-chained hot loop.
__global__ __launch_bounds__(128)
void gat_attention(const int* __restrict__ adj, const __bf16* __restrict__ hpB,
                   const float* __restrict__ s_src, const float* __restrict__ s_dst,
                   float* __restrict__ out) {
    int blk  = blockIdx.x;                 // BS*NHEAD*16 = 1024 blocks, 4 waves each
    int wave = threadIdx.x >> 5;
    int lane = threadIdx.x & 31;
    int rt   = (blk & 15) * 4 + wave;      // 16-row tile index (0..63)
    int head = (blk >> 4) & 7;
    int b    = blk >> 7;
    int bh   = b * NHEAD + head;

    int half = lane >> 4, mrow = lane & 15;
    int rowA = rt * 16 + mrow;             // row this lane owns in A-layout

    float ssrc = s_src[bh * NNODE + rowA];
    const float* sd     = s_dst + (size_t)bh * NNODE;
    const int*   adjrow = adj + ((size_t)(b * NNODE + rowA)) * NNODE;
    const v16bf* hb     = (const v16bf*)hpB + (size_t)bh * 32 * 4 * 32;

    // per-(b,h) max of d_j : 32 coalesced loads + wave reduce (L2 hits)
    float dm = -1e30f;
#pragma unroll
    for (int t = 0; t < 32; ++t) dm = fmaxf(dm, sd[t * 32 + lane]);
#pragma unroll
    for (int off = 1; off < 32; off <<= 1) dm = fmaxf(dm, __shfl_xor(dm, off, 32));

    float mshift = ssrc + dm;
    mshift = (mshift >= 0.f) ? mshift : NEG_SLOPE * mshift;   // row upper bound

    float lst = 0.f;                        // half-row partial denominator
    v8f acc0 = {0.f,0.f,0.f,0.f,0.f,0.f,0.f,0.f};
    v8f acc1 = acc0, acc2 = acc0, acc3 = acc0;

    for (int jt = 0; jt < 32; ++jt) {
        float p[16];
#pragma unroll
        for (int g = 0; g < 2; ++g) {
            int jb = jt * 32 + g * 16 + half * 8;
            float4 d0 = *(const float4*)(sd + jb);
            float4 d1 = *(const float4*)(sd + jb + 4);
            int4   a0 = *(const int4*)(adjrow + jb);
            int4   a1 = *(const int4*)(adjrow + jb + 4);
            float dv8[8] = {d0.x, d0.y, d0.z, d0.w, d1.x, d1.y, d1.z, d1.w};
            int   av8[8] = {a0.x, a0.y, a0.z, a0.w, a1.x, a1.y, a1.z, a1.w};
#pragma unroll
            for (int e = 0; e < 8; ++e) {
                float x = ssrc + dv8[e];
                x = (x >= 0.f) ? x : NEG_SLOPE * x;             // leaky_relu
                float pi = (av8[e] > 0) ? __expf(x - mshift) : 0.f;
                p[g * 8 + e] = pi;
                lst += pi;
            }
        }
        if (jt + 1 < 32) __builtin_prefetch(adjrow + (jt + 1) * 32, 0, 0);

        v16bf A;
#pragma unroll
        for (int i = 0; i < 16; ++i) A[i] = (__bf16)p[i];

        const v16bf* hbt = hb + (size_t)jt * 4 * 32;
        v16bf B0 = hbt[0 * 32 + lane];
        v16bf B1 = hbt[1 * 32 + lane];
        v16bf B2 = hbt[2 * 32 + lane];
        v16bf B3 = hbt[3 * 32 + lane];
        acc0 = __builtin_amdgcn_wmma_f32_16x16x32_bf16(false, A, false, B0, (short)0, acc0, false, false);
        acc1 = __builtin_amdgcn_wmma_f32_16x16x32_bf16(false, A, false, B1, (short)0, acc1, false, false);
        acc2 = __builtin_amdgcn_wmma_f32_16x16x32_bf16(false, A, false, B2, (short)0, acc2, false, false);
        acc3 = __builtin_amdgcn_wmma_f32_16x16x32_bf16(false, A, false, B3, (short)0, acc3, false, false);
    }

    // combine the two half-row partial sums -> full denominator per row
    lst += __shfl_xor(lst, 16, 32);

    // epilogue: divide by softmax denominator and store
#pragma unroll
    for (int r = 0; r < 8; ++r) {
        float lr  = __shfl(lst, r + 8 * half, 32);
        float inv = (lr > 0.f) ? 1.0f / lr : 0.f;
        int row = rt * 16 + r + 8 * half;
        float* orow = out + ((size_t)bh * NNODE + row) * FOUT;
        orow[0 * 16 + mrow] = acc0[r] * inv;
        orow[1 * 16 + mrow] = acc1[r] * inv;
        orow[2 * 16 + mrow] = acc2[r] * inv;
        orow[3 * 16 + mrow] = acc3[r] * inv;
    }
}

// ---------------- launcher ----------------
extern "C" void kernel_launch(void* const* d_in, const int* in_sizes, int n_in,
                              void* d_out, int out_size, void* d_ws, size_t ws_size,
                              hipStream_t stream) {
    (void)in_sizes; (void)n_in; (void)out_size; (void)ws_size;
    const float* h     = (const float*)d_in[0];
    const int*   adj   = (const int*)d_in[1];
    const float* w     = (const float*)d_in[2];
    const float* a_src = (const float*)d_in[3];
    const float* a_dst = (const float*)d_in[4];
    float* out = (float*)d_out;

    char* ws = (char*)d_ws;
    __bf16* hA    = (__bf16*)(ws + OFF_HA);
    __bf16* wB    = (__bf16*)(ws + OFF_WB);
    __bf16* hpB   = (__bf16*)(ws + OFF_HPB);
    float*  s_src = (float*)(ws + OFF_SSRC);
    float*  s_dst = (float*)(ws + OFF_SDST);

    // s_src/s_dst are contiguous: zero both (2 * 65536 floats)
    zero_f32<<<(131072 + 255) / 256, 256, 0, stream>>>(s_src, 131072);
    pack_hA<<<2097152 / 256, 256, 0, stream>>>(h, hA);
    pack_wB<<<131072 / 256, 256, 0, stream>>>(w, wB);
    gat_project<<<BS * NHEAD * (NNODE / 32), 256, 0, stream>>>(
        hA, wB, a_src, a_dst, hpB, s_src, s_dst);
    gat_attention<<<BS * NHEAD * 16, 128, 0, stream>>>(
        adj, hpB, s_src, s_dst, out);
}
